// chain_79362405695531
// MI455X (gfx1250) — matrix operand order repaired and checked by reading, not compiled, not run on hardware
//
#include <hip/hip_runtime.h>
#include <math.h>

// ---------------- problem constants ----------------
#define BB    2
#define CC    768
#define TT    9
#define PP    2809            // 53*53
#define NN    32
#define DIN   2841            // PP + NN
#define BT    16              // BB*(TT-1)
#define KPAD  2848            // DIN padded to multiple of 32
#define PPAD  2816            // PP padded to multiple of 16
#define OUT1  442368          // B*T*N*C
#define CNT_PER_CH 50562      // B*T*P
#define KSTEPS (KPAD / 32)    // 89
#define AROWB 80              // LDS row pitch (bytes) for 32 bf16 (64B) + bank-skew pad
#define ABUFB (64 * AROWB)    // one A stage buffer: 5120 B

typedef __attribute__((ext_vector_type(16))) __bf16 v16bf;
typedef __attribute__((ext_vector_type(8)))  float  v8f;

union V16 { unsigned u[8]; uint4 q[2]; v16bf v; };

static __device__ inline unsigned short f2bf(float f) {
    unsigned u = __float_as_uint(f);
    unsigned r = 0x7FFFu + ((u >> 16) & 1u);   // round-to-nearest-even
    return (unsigned short)((u + r) >> 16);
}

// CDNA5 async copy: global -> LDS, 16 bytes per lane, tracked by ASYNCcnt.
static __device__ inline void async_g2l_b128(unsigned lds_byte_addr, const void* gaddr) {
    asm volatile("global_load_async_to_lds_b128 %0, %1, off"
                 :: "v"(lds_byte_addr), "v"((unsigned long long)gaddr)
                 : "memory");
}
static __device__ inline void wait_async0() {
    asm volatile("s_wait_asynccnt 0x0" ::: "memory");
}

// ---------------- 1) BatchNorm statistics ----------------
__global__ __launch_bounds__(256) void bn_stats_kernel(
    const float* __restrict__ x, float* __restrict__ mean, float* __restrict__ rstd)
{
    const int c = blockIdx.x;
    const int tid = threadIdx.x;
    float s = 0.f, s2 = 0.f;
    for (int b = 0; b < BB; ++b) {
        const float* xp = x + ((size_t)(b * CC + c)) * (TT * PP);
        for (int i = tid; i < TT * PP; i += 256) {
            float v = xp[i];
            s += v; s2 += v * v;
        }
    }
    __shared__ float sh1[256], sh2[256];
    sh1[tid] = s; sh2[tid] = s2;
    __syncthreads();
    for (int st = 128; st > 0; st >>= 1) {
        if (tid < st) { sh1[tid] += sh1[tid + st]; sh2[tid] += sh2[tid + st]; }
        __syncthreads();
    }
    if (tid == 0) {
        float m = sh1[0] / (float)CNT_PER_CH;
        float var = sh2[0] / (float)CNT_PER_CH - m * m;
        mean[c] = m;
        rstd[c] = rsqrtf(var + 1e-5f);
    }
}

// ---------------- 2) weight fp32 -> padded bf16 ----------------
__global__ __launch_bounds__(256) void convert_w_kernel(
    const float* __restrict__ src, unsigned short* __restrict__ dst,
    int rows, int cols, int R, int LD)
{
    size_t idx = (size_t)blockIdx.x * 256 + threadIdx.x;
    size_t total = (size_t)R * LD;
    if (idx >= total) return;
    int j = (int)(idx % LD);
    int r = (int)(idx / LD);
    float v = (r < rows && j < cols) ? src[(size_t)r * cols + j] : 0.f;
    dst[idx] = f2bf(v);
}

// ---------------- 3) build kt (bf16, [bt][768][KPAD]) ----------------
__global__ __launch_bounds__(256) void build_kt_kernel(
    const float* __restrict__ tem, const float* __restrict__ x,
    const float* __restrict__ mean, const float* __restrict__ rstd,
    const float* __restrict__ gamma, const float* __restrict__ beta,
    unsigned short* __restrict__ kt)
{
    size_t idx = (size_t)blockIdx.x * 256 + threadIdx.x;
    const size_t total = (size_t)BT * CC * KPAD;
    if (idx >= total) return;
    int j  = (int)(idx % KPAD);
    size_t rc = idx / KPAD;
    int c  = (int)(rc % CC);
    int bt = (int)(rc / CC);
    int b0 = bt >> 3;
    int t  = (bt & 7) + 1;
    float v;
    if (j >= DIN) {
        v = 0.f;
    } else if (j < NN) {
        v = tem[((size_t)b0 * NN + j) * CC + c];
    } else {
        int p = j - NN;
        float xv = x[(((size_t)b0 * CC + c) * TT + t) * PP + p];
        v = (xv - mean[c]) * rstd[c] * gamma[c] + beta[c];
    }
    kt[idx] = f2bf(v);
}

// ---------------- 4) WMMA GEMM with async-LDS A staging ----------------
// Block = 8 waves, tile 64(M) x 256(N).  A panel (64x32 bf16) is DMA'd into LDS
// with global_load_async_to_lds_b128, double buffered; B fragments are
// register double-buffered.  A: bf16 [768][KPAD] per bt.  Bw: bf16 rows = out
// column, contiguous K (native fc_w layout).
// GELU==1: out = bf16 gelu(acc+bias), ldo=KPAD.  GELU==0: out = f32 acc+bias, ldo=PPAD.
template <int GELU>
__global__ __launch_bounds__(256) void gemm_wmma_kernel(
    const unsigned short* __restrict__ A,
    const unsigned short* __restrict__ Bw,
    const float* __restrict__ bias,
    unsigned short* __restrict__ outBF,
    float* __restrict__ outF,
    int nTiles, int nValid, int ldo)
{
    __shared__ unsigned char As[2 * ABUFB];

    const int bt   = blockIdx.z;
    const int tid  = threadIdx.x;
    const int lane = tid & 31;
    const int wave = tid >> 5;
    const int mr   = lane & 15;   // row (A) / column (B) within tile
    const int kh   = lane >> 4;   // lane half selects K sub-range
    const int mBase = blockIdx.y * 64;
    const int nt0   = (blockIdx.x * 8 + wave) * 2;

    A += (size_t)bt * CC * KPAD;

    // ---- async A-copy assignment: thread -> (row, 16B chunk) of 64x32 panel
    const int arow   = tid >> 2;         // 0..63
    const int achunk = tid & 3;          // 0..3 (16B each)
    const unsigned short* ag = A + (size_t)(mBase + arow) * KPAD + achunk * 8;
    const unsigned ldsBase = (unsigned)(size_t)(&As[0]) + (unsigned)(arow * AROWB + achunk * 16);

    const bool val0 = (nt0 < nTiles);
    const bool val1 = (nt0 + 1 < nTiles);
    const unsigned* Brow0 =
        (const unsigned*)(Bw + (size_t)((val0 ? nt0 : 0) * 16 + mr) * KPAD);
    const unsigned* Brow1 =
        (const unsigned*)(Bw + (size_t)((val1 ? nt0 + 1 : 0) * 16 + mr) * KPAD);

    v8f acc[4][2];
    #pragma unroll
    for (int mt = 0; mt < 4; ++mt) { acc[mt][0] = (v8f){}; acc[mt][1] = (v8f){}; }

    // B fragment loader: lane half kh holds K = kh*8 + [0..7] and 16 + kh*8 + [0..7]
    #define LOADB(dst, row, kk)                                   \
        do {                                                      \
            const unsigned* _p = (row) + ((kk) >> 1) + kh * 4;    \
            _Pragma("unroll")                                     \
            for (int _i = 0; _i < 8; ++_i) (dst).u[_i] = _p[_i + (_i & 4)];  \
        } while (0)

    // ---- prologue: stage 0 -> buffer 0, first B fragments
    async_g2l_b128(ldsBase, ag);
    V16 b0c, b1c;
    LOADB(b0c, Brow0, 0);
    LOADB(b1c, Brow1, 0);
    wait_async0();
    __syncthreads();

    for (int s = 0; s < KSTEPS; ++s) {
        const int k0  = s * 32;
        const int cur = s & 1;

        V16 b0n = b0c, b1n = b1c;
        if (s + 1 < KSTEPS) {
            // overlap: DMA next A panel + fetch next B while computing current
            async_g2l_b128(ldsBase + (cur ^ 1) * ABUFB,
                           (const char*)ag + (size_t)(k0 + 32) * 2);
            LOADB(b0n, Brow0, k0 + 32);
            LOADB(b1n, Brow1, k0 + 32);
        }

        const unsigned char* abuf = &As[cur * ABUFB];
        #pragma unroll
        for (int mt = 0; mt < 4; ++mt) {
            // A fragment (ISA 7.12.2): regs0-3 -> K=kh*8+[0..7] (bytes kh*16),
            // regs4-7 -> K=16+kh*8+[0..7] (bytes 32+kh*16)
            const unsigned char* arp = abuf + (mt * 16 + mr) * AROWB + kh * 16;
            V16 a;
            a.q[0] = *(const uint4*)(arp);
            a.q[1] = *(const uint4*)(arp + 32);
            acc[mt][0] = __builtin_amdgcn_wmma_f32_16x16x32_bf16(
                             false, a.v, false, b0c.v, (short)0, acc[mt][0], false, false);
            acc[mt][1] = __builtin_amdgcn_wmma_f32_16x16x32_bf16(
                             false, a.v, false, b1c.v, (short)0, acc[mt][1], false, false);
        }
        b0c = b0n;
        b1c = b1n;

        wait_async0();       // next buffer fully in LDS
        __syncthreads();     // all waves past reads of the buffer being refilled next
    }
    #undef LOADB

    if (GELU) outBF += (size_t)bt * CC * ldo;
    else      outF  += (size_t)bt * CC * ldo;

    // C/D layout: VGPR r -> M = r (lanes 0-15) / r+8 (lanes 16-31), N = lane&15
    #pragma unroll
    for (int mt = 0; mt < 4; ++mt) {
        #pragma unroll
        for (int tile = 0; tile < 2; ++tile) {
            bool valid = tile ? val1 : val0;
            if (!valid) continue;
            int n = (nt0 + tile) * 16 + mr;
            float bb = (n < nValid) ? bias[n] : 0.f;
            #pragma unroll
            for (int r = 0; r < 8; ++r) {
                int m = mBase + mt * 16 + kh * 8 + r;
                float v = acc[mt][tile][r] + bb;
                if (GELU) {
                    float g = 0.5f * v * (1.f + erff(v * 0.70710678118654752f));
                    outBF[(size_t)m * ldo + n] = f2bf(g);
                } else {
                    outF[(size_t)m * ldo + n] = v;
                }
            }
        }
    }
}

// ---------------- 5) score[bt][n][p] = sum_c q[n][c] * k[c][p] ----------------
__global__ __launch_bounds__(256) void score_kernel(
    const float* __restrict__ tem, const float* __restrict__ kbuf,
    float* __restrict__ score)
{
    const int bt = blockIdx.y;
    const int b0 = bt >> 3;
    const int p  = blockIdx.x * 256 + threadIdx.x;
    const bool valid = (p < PP);
    const float* kb = kbuf + (size_t)bt * CC * PPAD;

    __shared__ float qs[NN * 128];
    float acc[NN];
    #pragma unroll
    for (int n = 0; n < NN; ++n) acc[n] = 0.f;

    for (int c0 = 0; c0 < CC; c0 += 128) {
        for (int i = threadIdx.x; i < NN * 128; i += 256) {
            int n = i >> 7, cc = i & 127;
            qs[i] = tem[((size_t)b0 * NN + n) * CC + c0 + cc];
        }
        __syncthreads();
        if (valid) {
            for (int cc = 0; cc < 128; ++cc) {
                float kv = kb[(size_t)(c0 + cc) * PPAD + p];
                #pragma unroll
                for (int n = 0; n < NN; ++n) acc[n] += qs[(n << 7) + cc] * kv;
            }
        }
        __syncthreads();
    }
    if (valid) {
        #pragma unroll
        for (int n = 0; n < NN; ++n)
            score[((size_t)bt * NN + n) * PP + p] = acc[n];
    }
}

// ---------------- 6) top-1 (ties summed) gather + normalize-on-the-fly ----------------
__global__ __launch_bounds__(256) void gather_kernel(
    const float* __restrict__ x, const float* __restrict__ pos,
    const float* __restrict__ mean, const float* __restrict__ rstd,
    const float* __restrict__ gamma, const float* __restrict__ beta,
    const float* __restrict__ score, float* __restrict__ out)
{
    const int n  = blockIdx.x;
    const int bt = blockIdx.y;
    const int b0 = bt >> 3;
    const int t  = (bt & 7) + 1;
    const int tid = threadIdx.x;

    __shared__ float ss[PP];
    __shared__ float red[256];
    const float* srow = score + ((size_t)bt * NN + n) * PP;
    for (int p = tid; p < PP; p += 256) ss[p] = srow[p];
    __syncthreads();

    float lm = -3.4e38f;
    for (int p = tid; p < PP; p += 256) lm = fmaxf(lm, ss[p]);
    red[tid] = lm;
    __syncthreads();
    for (int st = 128; st > 0; st >>= 1) {
        if (tid < st) red[tid] = fmaxf(red[tid], red[tid + st]);
        __syncthreads();
    }
    const float mx = red[0];

    // deterministic serial tie-sum: each thread owns c = tid, tid+256, tid+512
    float accf[3] = {0.f, 0.f, 0.f};
    float accp[3] = {0.f, 0.f, 0.f};
    for (int p = 0; p < PP; ++p) {
        if (ss[p] == mx) {
            #pragma unroll
            for (int i = 0; i < 3; ++i) {
                int c = tid + 256 * i;
                size_t xi = (((size_t)b0 * CC + c) * TT + t) * PP + p;
                float xv = x[xi];
                accf[i] += (xv - mean[c]) * rstd[c] * gamma[c] + beta[c];
                accp[i] += pos[xi];
            }
        }
    }
    size_t ob = (((size_t)b0 * TT + t) * NN + n) * CC;
    #pragma unroll
    for (int i = 0; i < 3; ++i) {
        out[ob + tid + 256 * i]        = accf[i];
        out[OUT1 + ob + tid + 256 * i] = accp[i];
    }
}

// ---------------- 7) copy t=0 slots ----------------
__global__ __launch_bounds__(256) void copy_t0_kernel(
    const float* __restrict__ tem, const float* __restrict__ tem_pos,
    float* __restrict__ out)
{
    int idx = blockIdx.x * 256 + threadIdx.x;
    if (idx >= BB * NN * CC) return;
    int b = idx / (NN * CC);
    int r = idx % (NN * CC);
    size_t o = (size_t)b * TT * NN * CC + r;
    out[o]        = tem[idx];
    out[OUT1 + o] = tem_pos[idx];
}

// ---------------- launch ----------------
extern "C" void kernel_launch(void* const* d_in, const int* in_sizes, int n_in,
                              void* d_out, int out_size, void* d_ws, size_t ws_size,
                              hipStream_t stream) {
    const float* tem     = (const float*)d_in[0];
    const float* tem_pos = (const float*)d_in[1];
    const float* x       = (const float*)d_in[2];
    const float* pos     = (const float*)d_in[3];
    const float* gamma   = (const float*)d_in[4];
    const float* beta    = (const float*)d_in[5];
    const float* fc1_w   = (const float*)d_in[6];
    const float* fc1_b   = (const float*)d_in[7];
    const float* fc2_w   = (const float*)d_in[8];
    const float* fc2_b   = (const float*)d_in[9];
    float* out = (float*)d_out;

    // workspace layout (all offsets 256B aligned), total ~317 MB
    char* w = (char*)d_ws;
    size_t off = 0;
    float* mean = (float*)(w + off); off += 3072;
    float* rstd = (float*)(w + off); off += 3072;
    unsigned short* fc1wb = (unsigned short*)(w + off); off += (size_t)KPAD * KPAD * 2;
    unsigned short* fc2wb = (unsigned short*)(w + off); off += (size_t)PPAD * KPAD * 2;
    unsigned short* ktb   = (unsigned short*)(w + off); off += (size_t)BT * CC * KPAD * 2;
    unsigned short* hbuf  = (unsigned short*)(w + off); off += (size_t)BT * CC * KPAD * 2;
    float* kbuf  = (float*)(w + off); off += (size_t)BT * CC * PPAD * 4;
    float* score = (float*)(w + off); off += (size_t)BT * NN * PP * 4;

    // 1) batchnorm stats
    bn_stats_kernel<<<CC, 256, 0, stream>>>(x, mean, rstd);

    // 2) weights -> bf16 padded
    {
        size_t tot1 = (size_t)KPAD * KPAD;
        convert_w_kernel<<<(unsigned)((tot1 + 255) / 256), 256, 0, stream>>>(
            fc1_w, fc1wb, DIN, DIN, KPAD, KPAD);
        size_t tot2 = (size_t)PPAD * KPAD;
        convert_w_kernel<<<(unsigned)((tot2 + 255) / 256), 256, 0, stream>>>(
            fc2_w, fc2wb, PP, DIN, PPAD, KPAD);
    }

    // 3) build kt
    {
        size_t tot = (size_t)BT * CC * KPAD;
        build_kt_kernel<<<(unsigned)((tot + 255) / 256), 256, 0, stream>>>(
            tem, x, mean, rstd, gamma, beta, ktb);
    }

    // 4) GEMM1: h = gelu(kt @ fc1^T + b1)   [768 x 2841] per bt, bf16 out
    {
        int nTiles = KPAD / 16;                      // 178
        dim3 grid((nTiles + 15) / 16, CC / 64, BT);  // (12, 12, 16)
        gemm_wmma_kernel<1><<<grid, 256, 0, stream>>>(
            ktb, fc1wb, fc1_b, hbuf, nullptr, nTiles, DIN, KPAD);
    }

    // 5) GEMM2: k = h @ fc2^T + b2          [768 x 2809] per bt, f32 out
    {
        int nTiles = PPAD / 16;                      // 176
        dim3 grid((nTiles + 15) / 16, CC / 64, BT);  // (11, 12, 16)
        gemm_wmma_kernel<0><<<grid, 256, 0, stream>>>(
            hbuf, fc2wb, fc2_b, nullptr, kbuf, nTiles, PP, PPAD);
    }

    // 6) score = q . k
    score_kernel<<<dim3((PP + 255) / 256, BT), 256, 0, stream>>>(tem, kbuf, score);

    // 7) argmax + tie-sum gather (t = 1..8)
    gather_kernel<<<dim3(NN, BT), 256, 0, stream>>>(
        x, pos, mean, rstd, gamma, beta, score, out);

    // 8) t = 0 slots
    copy_t0_kernel<<<(BB * NN * CC + 255) / 256, 256, 0, stream>>>(tem, tem_pos, out);
}
